// Attention_26955214750248
// MI455X (gfx1250) — compile-verified, matrix operand order checked by reading, not run on hardware
//
#include <hip/hip_runtime.h>
#include <hip/hip_bf16.h>

#define DIM    768
#define HEADS  12
#define DHEAD  64
#define NSEQ   197
#define NPAD   224          // padded sequence rows for Q/K/Vt tiles
#define BATCH  64
#define NJT    13           // 13 j-tiles * 16 = 208 >= 197 (rows 208..223 stay zero)
#define MROWS  (BATCH * NSEQ)   // 12608
#define MTILES (MROWS / 16)     // 788
#define SZT    ((size_t)BATCH * HEADS * NPAD * DHEAD)   // elems per Q/K/Vt tensor

typedef __attribute__((ext_vector_type(16))) __bf16 v16bf;
typedef __attribute__((ext_vector_type(8)))  float  v8f;
typedef __attribute__((ext_vector_type(4)))  unsigned v4u;
typedef __attribute__((ext_vector_type(8)))  unsigned v8u;

union BFrag { v16bf v; __bf16 e[16]; uint4 q[2]; };
union FAcc  { v8f   v; float  e[8]; };
union H8    { __bf16 e[8]; uint4 q; };

// A/B 16-bit fragment, per-lane element K index: K = (e>>3)*16 + (lane>>4)*8 + (e&7)
// bf16 source, rows contiguous along K -> two 16B vector loads per lane
__device__ __forceinline__ v16bf frag_bf16_row(const __bf16* p, int ld, int lane) {
    const __bf16* r = p + (size_t)(lane & 15) * ld + ((lane >> 4) << 3);
    BFrag f;
    f.q[0] = *(const uint4*)(r);
    f.q[1] = *(const uint4*)(r + 16);
    return f.v;
}

__device__ __forceinline__ v8f wmma_bf16(v16bf a, v16bf b, v8f c) {
    return __builtin_amdgcn_wmma_f32_16x16x32_bf16(false, a, false, b, (short)0, c, false, false);
}

// ---- TDM: DMA `elems` contiguous bf16 from global into LDS at `lptr` -------
// D# group0: count=1, lds_addr = addr[31:0] of the shared buffer (generic LDS
//            pointers carry the LDS byte offset in their low 32 bits), 57b
//            global_addr, type=2 ("image")
// D# group1: data_size=2B, tensor_dim0=tile_dim0=elems, tensor_dim1=tile_dim1=1,
//            dim0_stride=elems  (i.e. a single-row 2D tile = flat copy)
// Passing the ptrtoint'd LDS address inside g0 also makes the shared buffer
// escape into the asm, so its loads cannot be folded to undef.
__device__ __forceinline__ void tdm_copy_to_lds(const void* gptr, void* lptr, unsigned elems) {
    unsigned long long ga = (unsigned long long)gptr;
    unsigned lds_off = (unsigned)(unsigned long long)lptr;   // LDS byte offset
    v4u g0 = { 1u, lds_off, (unsigned)ga,
               (unsigned)((ga >> 32) & 0x01FFFFFFu) | 0x80000000u };
    v8u g1 = { 0x00010000u,                               // data_size = 2 bytes
               (elems & 0xFFFFu) << 16,                   // tensor_dim0[15:0]
               ((elems >> 16) & 0xFFFFu) | (1u << 16),    // tensor_dim0[31:16], tensor_dim1=1
               (elems & 0xFFFFu) << 16,                   // tile_dim0
               1u,                                        // tile_dim1 = 1
               elems,                                     // tensor_dim0_stride[31:0]
               0u, 0u };
    asm volatile("tensor_load_to_lds %0, %1" :: "s"(g0), "s"(g1) : "memory");
}

__device__ __forceinline__ void wait_tensorcnt0() {
#if __has_builtin(__builtin_amdgcn_s_wait_tensorcnt)
    __builtin_amdgcn_s_wait_tensorcnt(0);
#else
    asm volatile("s_wait_tensorcnt 0x0" ::: "memory");
#endif
}

// ---- kernel 0: zero all of Q/K/Vt (guarantees pad rows/cols are 0) ---------
__global__ __launch_bounds__(256) void zero_qkv(__bf16* __restrict__ qkv) {
    size_t id = (size_t)blockIdx.x * blockDim.x + threadIdx.x;   // 8 elems each
    if (id >= 3 * SZT / 8) return;
    H8 z;
#pragma unroll
    for (int t = 0; t < 8; ++t) z.e[t] = (__bf16)0.f;
    *(uint4*)(qkv + id * 8) = z.q;
}

// ---- kernel 0b: x (fp32 row-major) -> xb (bf16 row-major) ------------------
__global__ __launch_bounds__(256) void cvt_x(const float* __restrict__ x, __bf16* __restrict__ xb) {
    size_t id = (size_t)blockIdx.x * blockDim.x + threadIdx.x;   // 8 elems each
    if (id >= (size_t)MROWS * DIM / 8) return;
    const float* p = x + id * 8;
    H8 o;
#pragma unroll
    for (int t = 0; t < 8; ++t) o.e[t] = (__bf16)p[t];
    *(uint4*)(xb + id * 8) = o.q;
}

// ---- kernel 0c: W[k][c] fp32 -> Wt[c][k] bf16 (both weights) ---------------
__global__ __launch_bounds__(256) void cvt_w(const float* __restrict__ W1,
                                             const float* __restrict__ W2,
                                             __bf16* __restrict__ W1t,
                                             __bf16* __restrict__ W2t) {
    const int n1 = (3 * DIM) * (DIM / 8);     // W_qkv output chunks
    const int n2 = DIM * (DIM / 8);           // W_out output chunks
    int id = blockIdx.x * blockDim.x + threadIdx.x;
    const float* W; __bf16* Wt; int cols; int idx;
    if (id < n1)            { W = W1; Wt = W1t; cols = 3 * DIM; idx = id; }
    else if (id < n1 + n2)  { W = W2; Wt = W2t; cols = DIM;     idx = id - n1; }
    else return;
    int c  = idx / (DIM / 8);
    int k8 = (idx % (DIM / 8)) * 8;
    H8 o;
#pragma unroll
    for (int t = 0; t < 8; ++t) o.e[t] = (__bf16)W[(size_t)(k8 + t) * cols + c];
    *(uint4*)(Wt + (size_t)c * DIM + k8) = o.q;
}

// ---- kernel 1: qkv = xb @ W1t^T, scatter Q/K row-major, V transposed -------
// A strip (16 x 768 bf16 = 24KB, contiguous) is TDM-staged into LDS once per
// block and shared by all 4 waves; B fragments stream from global/L2.
__global__ __launch_bounds__(128) void qkv_gemm(const __bf16* __restrict__ xb,
                                                const __bf16* __restrict__ W1t,
                                                __bf16* __restrict__ qkv) {
    const int lane  = threadIdx.x & 31;
    const int wave  = threadIdx.x >> 5;
    const int mtile = blockIdx.x;                       // 0..787
    const int cb    = blockIdx.y * 256 + wave * 64;     // column base (mult of 64)
    const __bf16* arow = xb  + (size_t)mtile * 16 * DIM;
    const __bf16* brow = W1t + (size_t)cb * DIM;        // rows = output columns

    __shared__ __bf16 atile[16 * DIM];
    if (threadIdx.x < 32) {
        tdm_copy_to_lds(arow, atile, 16 * DIM);
        wait_tensorcnt0();
    }
    __syncthreads();

    FAcc acc[4];
#pragma unroll
    for (int j = 0; j < 4; ++j)
#pragma unroll
        for (int i = 0; i < 8; ++i) acc[j].e[i] = 0.f;

    for (int k0 = 0; k0 < DIM; k0 += 32) {
        __builtin_prefetch(brow + k0 + 32, 0, 1);
        v16bf af = frag_bf16_row(atile + k0, DIM, lane);
#pragma unroll
        for (int j = 0; j < 4; ++j) {
            v16bf bf = frag_bf16_row(brow + (size_t)j * 16 * DIM + k0, DIM, lane);
            acc[j].v = wmma_bf16(af, bf, acc[j].v);
        }
    }

    const int which   = cb / DIM;                 // 0=Q 1=K 2=V(transposed)
    const int h       = (cb % DIM) / DHEAD;
    const int halfoff = (lane >> 4) << 3;
    __bf16* base = qkv + (size_t)which * SZT;
#pragma unroll
    for (int j = 0; j < 4; ++j) {
        const int d = j * 16 + (lane & 15);
#pragma unroll
        for (int r = 0; r < 8; ++r) {
            int gm = mtile * 16 + r + halfoff;
            int b  = gm / NSEQ;
            int n  = gm - b * NSEQ;
            size_t bh = (size_t)b * HEADS + h;
            if (which == 2)   // Vt[b][h][d][n]
                base[(bh * DHEAD + d) * NPAD + n] = (__bf16)acc[j].e[r];
            else              // Q/K[b][h][n][d]
                base[(bh * NPAD + n) * DHEAD + d] = (__bf16)acc[j].e[r];
        }
    }
}

// ---- kernel 2: per (b,h,i-tile) attention with softmax ---------------------
__global__ __launch_bounds__(32) void attn_kernel(const __bf16* __restrict__ qkv,
                                                  const float* __restrict__ scale,
                                                  __bf16* __restrict__ aout) {
    const int lane = threadIdx.x;
    int id = blockIdx.x;                 // b*H*13 + h*13 + it
    const int it = id % NJT;  id /= NJT;
    const int h  = id % HEADS; id /= HEADS;
    const int b  = id;
    const __bf16* Q  = qkv + ((size_t)b * HEADS + h) * NPAD * DHEAD;
    const __bf16* K  = Q + SZT;
    const __bf16* Vt = qkv + 2 * SZT + ((size_t)b * HEADS + h) * DHEAD * NPAD;
    const float sc = scale[h];
    const int halfoff = (lane >> 4) << 3;
    const int jcol    = lane & 15;

    // Q fragments for the 16-query tile, head dim split into two K=32 chunks
    v16bf qa0 = frag_bf16_row(Q + (size_t)it * 16 * DHEAD,      DHEAD, lane);
    v16bf qa1 = frag_bf16_row(Q + (size_t)it * 16 * DHEAD + 32, DHEAD, lane);

    // S = Q K^T over the whole (padded) key strip
    FAcc sacc[NJT];
#pragma unroll
    for (int jt = 0; jt < NJT; ++jt) {
        FAcc c;
#pragma unroll
        for (int i = 0; i < 8; ++i) c.e[i] = 0.f;
        v16bf kb0 = frag_bf16_row(K + (size_t)jt * 16 * DHEAD,      DHEAD, lane);
        v16bf kb1 = frag_bf16_row(K + (size_t)jt * 16 * DHEAD + 32, DHEAD, lane);
        c.v = wmma_bf16(qa0, kb0, c.v);
        c.v = wmma_bf16(qa1, kb1, c.v);
        sacc[jt] = c;
    }

    // scale, mask (j==i, j>=197), row softmax (max/sum across 16 lanes of a half)
    float rsum[8];
#pragma unroll
    for (int r = 0; r < 8; ++r) {
        const int i = it * 16 + r + halfoff;
        float m = -__builtin_inff();
#pragma unroll
        for (int jt = 0; jt < NJT; ++jt) {
            const int jj = jt * 16 + jcol;
            float s = sacc[jt].e[r] * sc;
            if (jj >= NSEQ || jj == i) s = -__builtin_inff();
            sacc[jt].e[r] = s;
            m = fmaxf(m, s);
        }
#pragma unroll
        for (int off = 1; off < 16; off <<= 1)
            m = fmaxf(m, __shfl_xor(m, off, 32));
        float sum = 0.f;
#pragma unroll
        for (int jt = 0; jt < NJT; ++jt) {
            float p = __expf(sacc[jt].e[r] - m);
            sacc[jt].e[r] = p;
            sum += p;
        }
#pragma unroll
        for (int off = 1; off < 16; off <<= 1)
            sum += __shfl_xor(sum, off, 32);
        rsum[r] = sum;
    }

    // Stage P through LDS to reshape C-layout -> A-fragments
    __shared__ float plds[16 * NPAD];
#pragma unroll
    for (int jt = 0; jt < NJT; ++jt)
#pragma unroll
        for (int r = 0; r < 8; ++r)
            plds[(r + halfoff) * NPAD + jt * 16 + jcol] = sacc[jt].e[r];
#pragma unroll
    for (int r = 0; r < 8; ++r)                     // zero tail j = 208..223
        plds[(r + halfoff) * NPAD + 208 + jcol] = 0.f;
    __syncthreads();

    // O = P @ V   (Vt rows are output columns d, contiguous along j)
    FAcc oacc[4];
#pragma unroll
    for (int dt = 0; dt < 4; ++dt)
#pragma unroll
        for (int i = 0; i < 8; ++i) oacc[dt].e[i] = 0.f;

#pragma unroll
    for (int j0 = 0; j0 < NPAD; j0 += 32) {
        BFrag pf;
#pragma unroll
        for (int g = 0; g < 2; ++g)
#pragma unroll
            for (int t = 0; t < 8; ++t)
                pf.e[g * 8 + t] = (__bf16)plds[(lane & 15) * NPAD + j0 + g * 16 + halfoff + t];
#pragma unroll
        for (int dt = 0; dt < 4; ++dt) {
            v16bf vb = frag_bf16_row(Vt + (size_t)dt * 16 * NPAD + j0, NPAD, lane);
            oacc[dt].v = wmma_bf16(pf.v, vb, oacc[dt].v);
        }
    }

    // normalize and store bf16 [b*NSEQ+n][h*64+d]
#pragma unroll
    for (int dt = 0; dt < 4; ++dt)
#pragma unroll
        for (int r = 0; r < 8; ++r) {
            const int n = it * 16 + r + halfoff;
            if (n < NSEQ) {
                float o = oacc[dt].e[r] / rsum[r];
                aout[((size_t)(b * NSEQ + n)) * DIM + h * DHEAD + dt * 16 + jcol] = (__bf16)o;
            }
        }
}

// ---- kernel 3: out = A @ W2t^T + b_out (fp32 result) -----------------------
__global__ __launch_bounds__(128) void out_gemm(const __bf16* __restrict__ A,
                                                const __bf16* __restrict__ W2t,
                                                const float* __restrict__ bout,
                                                float* __restrict__ out) {
    const int lane  = threadIdx.x & 31;
    const int wave  = threadIdx.x >> 5;
    const int mtile = blockIdx.x;
    const int cb    = blockIdx.y * 256 + wave * 64;
    const __bf16* arow = A   + (size_t)mtile * 16 * DIM;
    const __bf16* brow = W2t + (size_t)cb * DIM;

    __shared__ __bf16 atile[16 * DIM];
    if (threadIdx.x < 32) {
        tdm_copy_to_lds(arow, atile, 16 * DIM);
        wait_tensorcnt0();
    }
    __syncthreads();

    FAcc acc[4];
#pragma unroll
    for (int j = 0; j < 4; ++j)
#pragma unroll
        for (int i = 0; i < 8; ++i) acc[j].e[i] = 0.f;

    for (int k0 = 0; k0 < DIM; k0 += 32) {
        __builtin_prefetch(brow + k0 + 32, 0, 1);
        v16bf af = frag_bf16_row(atile + k0, DIM, lane);
#pragma unroll
        for (int j = 0; j < 4; ++j) {
            v16bf bf = frag_bf16_row(brow + (size_t)j * 16 * DIM + k0, DIM, lane);
            acc[j].v = wmma_bf16(af, bf, acc[j].v);
        }
    }

    const int halfoff = (lane >> 4) << 3;
#pragma unroll
    for (int j = 0; j < 4; ++j) {
        const int c = cb + j * 16 + (lane & 15);
        const float bb = bout[c];
#pragma unroll
        for (int r = 0; r < 8; ++r) {
            const int gm = mtile * 16 + r + halfoff;
            out[(size_t)gm * DIM + c] = acc[j].e[r] + bb;
        }
    }
}

// ---- launcher --------------------------------------------------------------
extern "C" void kernel_launch(void* const* d_in, const int* in_sizes, int n_in,
                              void* d_out, int out_size, void* d_ws, size_t ws_size,
                              hipStream_t stream) {
    const float* x     = (const float*)d_in[0];
    const float* scale = (const float*)d_in[1];
    const float* Wqkv  = (const float*)d_in[2];
    const float* Wout  = (const float*)d_in[3];
    const float* bout  = (const float*)d_in[4];
    float* out = (float*)d_out;

    // workspace layout (all bf16, every region a multiple of 16 bytes)
    __bf16* qkv  = (__bf16*)d_ws;                       // Q,K,Vt : 3*SZT
    __bf16* xb   = qkv + 3 * SZT;                       // MROWS*DIM
    __bf16* W1t  = xb  + (size_t)MROWS * DIM;           // 2304*768
    __bf16* W2t  = W1t + (size_t)(3 * DIM) * DIM;       // 768*768
    __bf16* aout = W2t + (size_t)DIM * DIM;             // MROWS*DIM

    {
        size_t n = 3 * SZT / 8;
        zero_qkv<<<(unsigned)((n + 255) / 256), 256, 0, stream>>>(qkv);
    }
    {
        size_t n = (size_t)MROWS * DIM / 8;
        cvt_x<<<(unsigned)((n + 255) / 256), 256, 0, stream>>>(x, xb);
    }
    {
        int n = (3 * DIM) * (DIM / 8) + DIM * (DIM / 8);
        cvt_w<<<(n + 255) / 256, 256, 0, stream>>>(Wqkv, Wout, W1t, W2t);
    }
    qkv_gemm<<<dim3(MTILES, 9), 128, 0, stream>>>(xb, W1t, qkv);
    attn_kernel<<<BATCH * HEADS * NJT, 32, 0, stream>>>(qkv, scale, aout);
    out_gemm<<<dim3(MTILES, 3), 128, 0, stream>>>(aout, W2t, bout, out);
}
